// LinearHopfieldASA_14817637171725
// MI455X (gfx1250) — matrix operand order repaired and checked by reading, not compiled, not run on hardware
//
#include <hip/hip_runtime.h>

// Problem constants (from reference setup_inputs)
#define BQ   4
#define SEQ  4096
#define DIM  1024
#define FEA  64
#define MTOT (BQ * SEQ)          // 16384 rows
#define EPSV 1e-6f

typedef __bf16 bf16_t;
typedef __attribute__((ext_vector_type(16))) __bf16 v16bf;
typedef __attribute__((ext_vector_type(8)))  float  v8f;
typedef __attribute__((ext_vector_type(4)))  unsigned int uint4v;

static __device__ __forceinline__ v8f zero8() {
  v8f z;
#pragma unroll
  for (int i = 0; i < 8; ++i) z[i] = 0.f;
  return z;
}

// Async global->LDS copy of 16 bytes (CDNA5 async engine, ASYNCcnt).
// LDS operand is the LDS byte offset (= low 32 bits of the flat address).
static __device__ __forceinline__ void async_ld16(unsigned lds_off, const void* g) {
  asm volatile("global_load_async_to_lds_b128 %0, %1, off"
               :: "v"(lds_off), "v"(g) : "memory");
}
static __device__ __forceinline__ void wait_async0() {
  asm volatile("s_wait_asynccnt 0x0" ::: "memory");
}
static __device__ __forceinline__ unsigned lds_off_of(const void* p) {
  return (unsigned)(size_t)p;
}

// A-fragment (16x32 bf16, MxK): lane l holds row (l&15); K chunks at
// k0 + (l>>4)*8 (VGPR0-3) and k0 + 16 + (l>>4)*8 (VGPR4-7). Row-major source.
static __device__ __forceinline__ v16bf load_frag_a(const bf16_t* base, int ld,
                                                    int m0, int k0) {
  const int lane = threadIdx.x & 31;
  const int r = lane & 15, h = lane >> 4;
  const bf16_t* p = base + (size_t)(m0 + r) * ld + k0 + h * 8;
  union { v16bf v; uint4v u[2]; } t;
  t.u[0] = *(const uint4v*)p;
  t.u[1] = *(const uint4v*)(p + 16);
  return t.v;
}

// B-fragment (32x16 bf16, KxN): lane l holds column (l&15); K chunks at
// k0 + (l>>4)*8 (VGPR0-3) and k0 + 16 + (l>>4)*8 (VGPR4-7). Source layout: row n of baseT (leading
// dim ld along K) is column n of the logical B matrix.
static __device__ __forceinline__ v16bf load_frag_b(const bf16_t* baseT, int ld,
                                                    int n0, int k0) {
  const int lane = threadIdx.x & 31;
  const int c = lane & 15, h = lane >> 4;
  const bf16_t* p = baseT + (size_t)(n0 + c) * ld + k0 + h * 8;
  union { v16bf v; uint4v u[2]; } t;
  t.u[0] = ((const uint4v*)p)[0];
  t.u[1] = *(const uint4v*)(p + 16);
  return t.v;
}

static __device__ __forceinline__ v8f wmma_bf16(v16bf a, v16bf b, v8f c) {
  // (neg_a, A, neg_b, B, c_mod, C, reuse_a, reuse_b)
  return __builtin_amdgcn_wmma_f32_16x16x32_bf16(false, a, false, b, (short)0,
                                                 c, false, false);
}

// ---------------- prep kernels ----------------

__global__ void k_cvt_bf16(const float* __restrict__ src,
                           bf16_t* __restrict__ dst, long n) {
  long i = (long)blockIdx.x * blockDim.x + threadIdx.x;
  if (i < n) dst[i] = (bf16_t)src[i];
}

// Wt[n][k] = W[k][n] as bf16
__global__ void k_transpose_bf16(const float* __restrict__ W,
                                 bf16_t* __restrict__ Wt, int K, int Nn) {
  long i = (long)blockIdx.x * blockDim.x + threadIdx.x;
  long n = (long)K * Nn;
  if (i >= n) return;
  int k = (int)(i / Nn), c = (int)(i % Nn);
  Wt[(size_t)c * K + k] = (bf16_t)W[i];
}

// ---------------- LDS-staged bf16 WMMA GEMM ----------------
// C[M,Nn] = act(A[M,K] @ W[K,Nn] + bias), W given pre-transposed as Wt[Nn,K].
// Block = 256 threads (8 waves) computing a 128(M) x 64(N) tile. A and B
// k-slices are staged into LDS with async global->LDS copies, double-buffered
// so the copy of slice k+1 overlaps the 4 WMMAs per wave on slice k. The
// 64x32 weight slice is fetched once per block and shared by all 8 waves.
__global__ void k_gemm(const bf16_t* __restrict__ A,
                       const bf16_t* __restrict__ Wt,
                       const float* __restrict__ bias,
                       int M, int K, int Nn, int act,
                       bf16_t* __restrict__ outR,   // row-major bf16 or null
                       float* __restrict__ outF,    // row-major f32 or null
                       bf16_t* __restrict__ outT,   // [batch][Nn][Nb] bf16 or null
                       int Nb) {
  const int tid = threadIdx.x, wid = tid >> 5, lane = tid & 31;
  const int tiles_n = Nn >> 6;
  const int bm = (int)blockIdx.x / tiles_n;
  const int m0 = bm << 7;                  // 128-row block
  if (m0 >= M) return;
  const int n0 = ((int)blockIdx.x - bm * tiles_n) << 6;

  __shared__ __align__(16) bf16_t Ab[2][128 * 32];  // [m][k] slice
  __shared__ __align__(16) bf16_t Bb[2][64 * 32];   // [n][k] slice (transposed W)

  // Stage one 32-wide k-slice of A (128x32) and B (64x32) into buffer p.
  auto stage = [&](int p, int k0) {
#pragma unroll
    for (int r = 0; r < 2; ++r) {                   // A: 512 x 16B chunks
      const int c = tid + r * 256;
      const int m = c >> 2, kk = (c & 3) * 8;
      async_ld16(lds_off_of(&Ab[p][m * 32 + kk]),
                 A + (size_t)(m0 + m) * K + k0 + kk);
    }
    {                                               // B: 256 x 16B chunks
      const int n = tid >> 2, kk = (tid & 3) * 8;
      async_ld16(lds_off_of(&Bb[p][n * 32 + kk]),
                 Wt + (size_t)(n0 + n) * K + k0 + kk);
    }
  };

  stage(0, 0);
  wait_async0();
  __syncthreads();

  v8f acc0 = zero8(), acc1 = zero8(), acc2 = zero8(), acc3 = zero8();
  int p = 0;
  for (int k0 = 0; k0 < K; k0 += 32) {
    if (k0 + 32 < K) stage(p ^ 1, k0 + 32);         // overlap copy with compute
    v16bf a = load_frag_a(&Ab[p][0], 32, wid * 16, 0);
    acc0 = wmma_bf16(a, load_frag_b(&Bb[p][0], 32,  0, 0), acc0);
    acc1 = wmma_bf16(a, load_frag_b(&Bb[p][0], 32, 16, 0), acc1);
    acc2 = wmma_bf16(a, load_frag_b(&Bb[p][0], 32, 32, 0), acc2);
    acc3 = wmma_bf16(a, load_frag_b(&Bb[p][0], 32, 48, 0), acc3);
    wait_async0();
    __syncthreads();
    p ^= 1;
  }

  const int rh = (lane >> 4) * 8, nc = lane & 15;
  v8f accs[4] = {acc0, acc1, acc2, acc3};
#pragma unroll
  for (int s = 0; s < 4; ++s) {
#pragma unroll
    for (int g = 0; g < 8; ++g) {
      const int row = m0 + wid * 16 + g + rh;
      const int col = n0 + s * 16 + nc;
      float v = accs[s][g] + bias[col];
      if (act) v = (v > 0.f) ? (v + 1.f) : __expf(v);   // elu(v)+1
      if (outR) outR[(size_t)row * Nn + col] = (bf16_t)v;
      if (outF) outF[(size_t)row * Nn + col] = v;
      if (outT) {
        const int bb = row / Nb, nl = row - bb * Nb;
        outT[((size_t)bb * Nn + col) * Nb + nl] = (bf16_t)v;
      }
    }
  }
}

// ---------------- chunked causal linear attention ----------------
// Grid: (D/64 slices, B batches). Block: 256 threads = 8 waves.
// Per chunk (64 rows): stage Qc,Kc,Vc,Kt_c into LDS (async), then
// S = Qc Kc^T (masked) ; out = S_m Vc + Qc KV ; den from masked rowsums;
// KV^T += Vc^T Kc ; Ks += colsum(Kc).
__global__ void k_linattn(const bf16_t* __restrict__ Qn,  // [B*SEQ, FEA]
                          const bf16_t* __restrict__ Kn,  // [B*SEQ, FEA]
                          const bf16_t* __restrict__ Kt,  // [B][FEA][SEQ]
                          const bf16_t* __restrict__ Vt,  // [B][DIM][SEQ]
                          bf16_t* __restrict__ Oat) {     // [B*SEQ, DIM]
  const int slice = blockIdx.x;            // 0..15 (64 cols of D each)
  const int b     = blockIdx.y;            // 0..3
  const int d0    = slice * 64;
  const int tid   = threadIdx.x;
  const int wid   = tid >> 5;
  const int lane  = tid & 31;

  __shared__ __align__(16) bf16_t Smb[64 * 64];   // masked S (bf16)
  __shared__ __align__(16) float  KVf[64 * 64];   // KV^T f32  [d][f]
  __shared__ __align__(16) bf16_t KVb[64 * 64];   // KV^T bf16 [d][f]
  __shared__ __align__(16) bf16_t QcS[64 * 64];   // Q chunk  [i][f]
  __shared__ __align__(16) bf16_t KcS[64 * 64];   // K chunk  [j][f]
  __shared__ __align__(16) bf16_t VcS[64 * 64];   // V^T chunk [d][j]
  __shared__ __align__(16) bf16_t KtS[64 * 64];   // K^T chunk [f][j]
  __shared__ float Ks[64], csum[64], den[64];

  for (int i = tid; i < 64 * 64; i += 256) { KVf[i] = 0.f; KVb[i] = (bf16_t)0.f; }
  if (tid < 64) Ks[tid] = 0.f;
  __syncthreads();

  const bf16_t* Qb  = Qn + (size_t)b * SEQ * FEA;
  const bf16_t* Kb  = Kn + (size_t)b * SEQ * FEA;
  const bf16_t* Ktb = Kt + (size_t)b * FEA * SEQ;
  const bf16_t* Vtb = Vt + ((size_t)b * DIM + d0) * SEQ;  // 64 rows, ld=SEQ

  for (int c = 0; c < SEQ / 64; ++c) {
    const int n0 = c * 64;
    const bf16_t* Qc = Qb + (size_t)n0 * FEA;  // [64 x FEA], ld FEA
    const bf16_t* Kc = Kb + (size_t)n0 * FEA;

    // --- Phase 0: async-stage this chunk's tiles into LDS ---
#pragma unroll
    for (int r = 0; r < 2; ++r) {
      const int cc = tid + r * 256;            // 512 x 16B chunks per array
      const int m = cc >> 3, kk = (cc & 7) * 8;
      async_ld16(lds_off_of(&QcS[m * 64 + kk]), Qc + m * 64 + kk);
      async_ld16(lds_off_of(&KcS[m * 64 + kk]), Kc + m * 64 + kk);
      async_ld16(lds_off_of(&VcS[m * 64 + kk]), Vtb + (size_t)m * SEQ + n0 + kk);
      async_ld16(lds_off_of(&KtS[m * 64 + kk]), Ktb + (size_t)m * SEQ + n0 + kk);
    }
    wait_async0();
    __syncthreads();

    // --- Phase 1: S = Qc @ Kc^T, causal mask (i>=j), store bf16 ---
#pragma unroll
    for (int t = 0; t < 2; ++t) {
      const int tt = wid * 2 + t;
      const int ti = tt >> 2, tj = tt & 3;
      v8f acc = zero8();
      for (int k0 = 0; k0 < FEA; k0 += 32) {
        v16bf a  = load_frag_a(QcS, FEA, ti * 16, k0);
        v16bf bm = load_frag_b(KcS, FEA, tj * 16, k0);  // col j = row j of Kc
        acc = wmma_bf16(a, bm, acc);
      }
      const int rh = (lane >> 4) * 8, nc = lane & 15;
#pragma unroll
      for (int g = 0; g < 8; ++g) {
        const int i = ti * 16 + g + rh, j = tj * 16 + nc;
        Smb[i * 64 + j] = (bf16_t)((i >= j) ? acc[g] : 0.f);
      }
    }
    __syncthreads();

    // --- Phase 1.5: denominator and K column-sums (from LDS) ---
    if (tid < 64) {
      const int i = tid;
      float rs = 0.f;
      for (int j = 0; j <= i; ++j) rs += (float)Smb[i * 64 + j];
      float qk = 0.f;
      for (int f = 0; f < FEA; ++f) qk += (float)QcS[i * FEA + f] * Ks[f];
      den[i] = rs + qk + EPSV;
    } else if (tid < 128) {
      const int f = tid - 64;
      float s = 0.f;
      for (int j = 0; j < 64; ++j) s += (float)KcS[j * FEA + f];
      csum[f] = s;
    }
    __syncthreads();

    // --- Phase 2: out = (S_m @ Vc + Qc @ KV_prev) / den ---
#pragma unroll
    for (int t = 0; t < 2; ++t) {
      const int tt = wid * 2 + t;
      const int ti = tt >> 2, td = tt & 3;
      v8f acc = zero8();
      for (int k0 = 0; k0 < 64; k0 += 32) {          // intra-chunk: S_m @ Vc
        v16bf a  = load_frag_a(Smb, 64, ti * 16, k0);
        v16bf bm = load_frag_b(VcS, 64, td * 16, k0);
        acc = wmma_bf16(a, bm, acc);
      }
      for (int k0 = 0; k0 < FEA; k0 += 32) {         // inter-chunk: Qc @ KV
        v16bf a  = load_frag_a(QcS, FEA, ti * 16, k0);
        v16bf bm = load_frag_b(KVb, 64, td * 16, k0);
        acc = wmma_bf16(a, bm, acc);
      }
      const int rh = (lane >> 4) * 8, nc = lane & 15;
#pragma unroll
      for (int g = 0; g < 8; ++g) {
        const int i = ti * 16 + g + rh, dd = td * 16 + nc;
        const float v = acc[g] / den[i];
        Oat[((size_t)b * SEQ + n0 + i) * DIM + d0 + dd] = (bf16_t)v;
      }
    }
    __syncthreads();

    // --- Phase 3: KV^T += Vc^T @ Kc ; Ks += csum ---
#pragma unroll
    for (int t = 0; t < 2; ++t) {
      const int tt = wid * 2 + t;
      const int td = tt >> 2, tf = tt & 3;
      v8f acc = zero8();
      for (int k0 = 0; k0 < 64; k0 += 32) {
        v16bf a  = load_frag_a(VcS, 64, td * 16, k0);   // rows d, red j
        v16bf bm = load_frag_b(KtS, 64, tf * 16, k0);   // col f = KtS row f
        acc = wmma_bf16(a, bm, acc);
      }
      const int rh = (lane >> 4) * 8, nc = lane & 15;
#pragma unroll
      for (int g = 0; g < 8; ++g) {
        const int dd = td * 16 + g + rh, f = tf * 16 + nc;
        const float nv = KVf[dd * 64 + f] + acc[g];
        KVf[dd * 64 + f] = nv;
        KVb[dd * 64 + f] = (bf16_t)nv;
      }
    }
    if (tid < 64) Ks[tid] += csum[tid];
    __syncthreads();
  }
}

// ---------------- host launch ----------------

extern "C" void kernel_launch(void* const* d_in, const int* in_sizes, int n_in,
                              void* d_out, int out_size, void* d_ws, size_t ws_size,
                              hipStream_t stream) {
  const float* x  = (const float*)d_in[0];
  const float* Wq = (const float*)d_in[1];
  const float* bq = (const float*)d_in[2];
  const float* Wk = (const float*)d_in[3];
  const float* bk = (const float*)d_in[4];
  const float* Wv = (const float*)d_in[5];
  const float* bv = (const float*)d_in[6];
  const float* Wo = (const float*)d_in[7];
  const float* bo = (const float*)d_in[8];
  float* out = (float*)d_out;
  (void)in_sizes; (void)n_in; (void)out_size; (void)ws_size;

  char* ws = (char*)d_ws;
  size_t off = 0;
  auto alloc = [&](size_t bytes) -> void* {
    void* p = ws + off;
    off += (bytes + 255) & ~(size_t)255;
    return p;
  };
  bf16_t* xb  = (bf16_t*)alloc((size_t)MTOT * DIM * 2);
  bf16_t* Wqt = (bf16_t*)alloc((size_t)FEA * DIM * 2);
  bf16_t* Wkt = (bf16_t*)alloc((size_t)FEA * DIM * 2);
  bf16_t* Wvt = (bf16_t*)alloc((size_t)DIM * DIM * 2);
  bf16_t* Wot = (bf16_t*)alloc((size_t)DIM * DIM * 2);
  bf16_t* Qn  = (bf16_t*)alloc((size_t)MTOT * FEA * 2);
  bf16_t* Kn  = (bf16_t*)alloc((size_t)MTOT * FEA * 2);
  bf16_t* Kt  = (bf16_t*)alloc((size_t)BQ * FEA * SEQ * 2);
  bf16_t* Vt  = (bf16_t*)alloc((size_t)BQ * DIM * SEQ * 2);
  bf16_t* Oat = (bf16_t*)alloc((size_t)MTOT * DIM * 2);

  // 1) convert x to bf16
  {
    long n = (long)MTOT * DIM;
    k_cvt_bf16<<<(unsigned)((n + 255) / 256), 256, 0, stream>>>(x, xb, n);
  }
  // 2) transpose+convert weights
  k_transpose_bf16<<<(DIM * FEA + 255) / 256, 256, 0, stream>>>(Wq, Wqt, DIM, FEA);
  k_transpose_bf16<<<(DIM * FEA + 255) / 256, 256, 0, stream>>>(Wk, Wkt, DIM, FEA);
  k_transpose_bf16<<<(DIM * DIM + 255) / 256, 256, 0, stream>>>(Wv, Wvt, DIM, DIM);
  k_transpose_bf16<<<(DIM * DIM + 255) / 256, 256, 0, stream>>>(Wo, Wot, DIM, DIM);

  // 3) Q = elu(x Wq + bq)+1 -> Qn (row-major bf16)
  k_gemm<<<(MTOT / 128) * (FEA / 64), 256, 0, stream>>>(
      xb, Wqt, bq, MTOT, DIM, FEA, 1, Qn, nullptr, nullptr, SEQ);
  // 4) K = elu(x Wk + bk)+1 -> Kn + transposed Kt
  k_gemm<<<(MTOT / 128) * (FEA / 64), 256, 0, stream>>>(
      xb, Wkt, bk, MTOT, DIM, FEA, 1, Kn, nullptr, Kt, SEQ);
  // 5) V = x Wv + bv -> transposed Vt only
  k_gemm<<<(MTOT / 128) * (DIM / 64), 256, 0, stream>>>(
      xb, Wvt, bv, MTOT, DIM, DIM, 0, nullptr, nullptr, Vt, SEQ);
  // 6) chunked causal linear attention -> Oat (bf16 row-major)
  {
    dim3 grid(DIM / 64, BQ);
    k_linattn<<<grid, 256, 0, stream>>>(Qn, Kn, Kt, Vt, Oat);
  }
  // 7) y = Oat @ Wo + bo -> f32 d_out
  k_gemm<<<(MTOT / 128) * (DIM / 64), 256, 0, stream>>>(
      Oat, Wot, bo, MTOT, DIM, DIM, 0, nullptr, out, nullptr, SEQ);
}
